// MultiHeadSelfAttention_22814866276576
// MI455X (gfx1250) — matrix operand order repaired and checked by reading, not compiled, not run on hardware
//
#include <hip/hip_runtime.h>

// MHA forward: B=4, S=2048, D=1024, H=16, HD=64.
// GEMMs + attention: TDM (tensor_load_to_lds) streams operand tiles into
// double-buffered LDS (s_wait_tensorcnt + one workgroup barrier per k-step);
// waves consume via ds_load_b128 into v_wmma_f32_16x16x32_f16.
// Attention is flash-style (online softmax), one wave per 16-query tile,
// 4 waves/block sharing one (b,h)'s K/V tiles.

#define BB 4
#define SS 2048
#define DD 1024
#define HH 16
#define HD 64

typedef __attribute__((ext_vector_type(16))) _Float16 v16h;
typedef __attribute__((ext_vector_type(8)))  float    v8f;
typedef unsigned int v4u __attribute__((ext_vector_type(4)));
typedef int          v8i __attribute__((ext_vector_type(8)));
typedef int          v4i __attribute__((ext_vector_type(4)));

#if __has_include(<hip/amd_detail/amd_gfx1250_TDM.h>)
#define TDM_6ARG 1
#endif

union FragU { v16h h; uint4 u[2]; };

// ---- WMMA fragment loaders -------------------------------------------------
// A-matrix 16x32 f16 (ISA 7.12.2): lane l holds row (l&15);
//   halfs 0-7  = K { l<16 ? 0..7 : 8..15 } + k0 ; halfs 8-15 = same + 16
__device__ __forceinline__ v16h load_fragA(const _Float16* __restrict__ base,
                                           int ld, int row, int k0, int lane) {
  const _Float16* p = base + (size_t)row * ld + k0 + ((lane & 16) ? 8 : 0);
  FragU f;
  f.u[0] = *reinterpret_cast<const uint4*>(p);
  f.u[1] = *reinterpret_cast<const uint4*>(p + 16);
  return f.h;
}

// B-matrix 32x16 f16: lane l holds column (l&15);
//   halfs 0-7 = K { l<16 ? 0..7 : 8..15 } + k0 ; halfs 8-15 = same + 16 (K-major source)
__device__ __forceinline__ v16h load_fragB(const _Float16* base,
                                           int ld, int col, int k0, int lane) {
  const _Float16* p = base + (size_t)col * ld + k0 + ((lane & 16) ? 8 : 0);
  FragU f;
  f.u[0] = *reinterpret_cast<const uint4*>(p);
  f.u[1] = *reinterpret_cast<const uint4*>(p + 16);
  return f.h;
}

__device__ __forceinline__ v8f wmma16(v16h a, v16h b, v8f c) {
  return __builtin_amdgcn_wmma_f32_16x16x32_f16(false, a, false, b,
                                                (short)0, c, false, false);
}

// ---- TDM: 2-D tile of 16-bit elements, global -> LDS -----------------------
// D# per CDNA5 ISA. dim0 = contiguous axis.
__device__ __forceinline__ void tdm_load_2d(unsigned lds_off, const void* gptr,
                                            unsigned tile_d0, unsigned tile_d1,
                                            unsigned tensor_d0, unsigned tensor_d1,
                                            unsigned stride_d0) {
  unsigned long long ga = (unsigned long long)(uintptr_t)gptr;
  v4u g0;
  g0.x = 1u;                                        // count=1 (valid, user mode)
  g0.y = lds_off;                                   // lds_addr [63:32]
  g0.z = (unsigned)(ga & 0xffffffffu);              // global_addr [95:64]
  g0.w = (unsigned)((ga >> 32) & 0x01ffffffu) | (2u << 30); // addr hi | type=2
  v8i g1;
  g1[0] = (int)(1u << 16);                          // data_size=1 -> 2 bytes
  g1[1] = (int)((tensor_d0 & 0xffffu) << 16);       // tensor_dim0[15:0]
  g1[2] = (int)((tensor_d0 >> 16) | ((tensor_d1 & 0xffffu) << 16));
  g1[3] = (int)((tensor_d1 >> 16) | ((tile_d0 & 0xffffu) << 16)); // tile_dim0
  g1[4] = (int)(tile_d1 & 0xffffu);                 // tile_dim1 (tile_dim2=0)
  g1[5] = (int)stride_d0;                           // tensor_dim0_stride lo
  g1[6] = 0;
  g1[7] = 0;
  v4i z4 = {0, 0, 0, 0};
#ifdef TDM_6ARG
  v8i z8 = {0, 0, 0, 0, 0, 0, 0, 0};
  __builtin_amdgcn_tensor_load_to_lds(g0, g1, z4, z4, z8, 0);
#else
  __builtin_amdgcn_tensor_load_to_lds(g0, g1, z4, z4, 0);
#endif
}

// ---- conversion kernels ----------------------------------------------------
__global__ void __launch_bounds__(256)
cvt_f32_f16(const float* __restrict__ src, _Float16* __restrict__ dst, int n) {
  int i = blockIdx.x * 256 + threadIdx.x;
  if (i < n) dst[i] = (_Float16)src[i];
}

// wt[n*rows + k] = w[k*cols + n]   (row-major [K,N] -> K-major [N,K], f16)
__global__ void __launch_bounds__(256)
transpose_cvt(const float* __restrict__ w, _Float16* __restrict__ wt,
              int rows, int cols) {
  int i = blockIdx.x * 256 + threadIdx.x;
  if (i < rows * cols) {
    int n = i % cols, k = i / cols;
    wt[(size_t)n * rows + k] = (_Float16)w[(size_t)k * cols + n];
  }
}

// ---- cooperative GEMM mainloop: 8 waves, 256x64 macro-tile ------------------
// B tile (64 cols x 32 k, f16, K-major) double-buffered in LDS via TDM.
// One barrier per k-step: it both publishes buf[cur] (after wave0's
// s_wait_tensorcnt) and proves all ds reads of buf[cur^1] retired before the
// next TDM overwrites it.
__device__ __forceinline__ void gemm_mainloop(
    const _Float16* __restrict__ ah, const _Float16* __restrict__ wt,
    _Float16 (*bsh)[64 * 32], int rowbase, int n0, int Ntot,
    int wave, int lane, v8f c0[4], v8f c1[4]) {
  const int arow0 = rowbase + (lane & 15);
  const int arow1 = arow0 + 16;
  if (wave == 0)
    tdm_load_2d((unsigned)(uintptr_t)&bsh[0][0], wt + (size_t)n0 * DD,
                32, 64, DD, (unsigned)Ntot, DD);
  for (int k0 = 0; k0 < DD; k0 += 32) {
    const int cur = (k0 >> 5) & 1;
    if (wave == 0) __builtin_amdgcn_s_wait_tensorcnt(0);  // buf[cur] landed
    __syncthreads();
    if (wave == 0 && (k0 + 32) < DD)
      tdm_load_2d((unsigned)(uintptr_t)&bsh[cur ^ 1][0],
                  wt + (size_t)n0 * DD + (k0 + 32), 32, 64, DD,
                  (unsigned)Ntot, DD);
    v16h a0 = load_fragA(ah, DD, arow0, k0, lane);
    v16h a1 = load_fragA(ah, DD, arow1, k0, lane);
#pragma unroll
    for (int t = 0; t < 4; ++t) {
      v16h b = load_fragB(&bsh[cur][0], 32, t * 16 + (lane & 15), 0, lane);
      c0[t] = wmma16(a0, b, c0[t]);
      c1[t] = wmma16(a1, b, c1[t]);
    }
  }
}

// ---- QKV projection: [B*S,3D] = x_h @ w_qkv + bias -------------------------
__global__ void __launch_bounds__(256)
gemm_qkv(const _Float16* __restrict__ xh, const _Float16* __restrict__ wt,
         const float* __restrict__ bias, _Float16* __restrict__ qh,
         _Float16* __restrict__ kh, _Float16* __restrict__ vt) {
  __shared__ __align__(16) _Float16 bsh[2][64 * 32];
  const int wave = threadIdx.x >> 5, lane = threadIdx.x & 31;
  const int nt = blockIdx.x % 48, mb = blockIdx.x / 48;   // 32 x 48 blocks
  const int rowbase = mb * 256 + wave * 32;
  const int n0 = nt * 64;

  v8f c0[4] = {}, c1[4] = {};
  gemm_mainloop(xh, wt, bsh, rowbase, n0, 3 * DD, wave, lane, c0, c1);

  const int hi = (lane & 16) ? 8 : 0;
#pragma unroll
  for (int t = 0; t < 4; ++t) {
    const int col = n0 + t * 16 + (lane & 15);
    const int which = col >> 10;                    // 0=Q,1=K,2=V
    const int rem = col & 1023, h = rem >> 6, hd = rem & 63;
    const float bv = bias[col];
#pragma unroll
    for (int g = 0; g < 2; ++g) {
      const v8f* cc = g ? c1 : c0;
#pragma unroll
      for (int j = 0; j < 8; ++j) {
        const int m = rowbase + g * 16 + j + hi;    // row in B*S
        const int b = m >> 11, s = m & 2047;
        const float acc = cc[t][j] + bv;
        const size_t bh = (size_t)(b * HH + h);
        if (which == 0)      qh[(bh * SS + s) * HD + hd] = (_Float16)(acc * 0.125f);
        else if (which == 1) kh[(bh * SS + s) * HD + hd] = (_Float16)acc;
        else                 vt[(bh * HD + hd) * SS + s] = (_Float16)acc;
      }
    }
  }
}

// ---- flash attention --------------------------------------------------------
// Block = 4 waves, one (b,h), 4 consecutive 16-query tiles. K/V tiles for 32
// keys are TDM-streamed into double-buffered LDS shared by all 4 waves.
//   kbuf: [32 keys][64 hd]  (hd contiguous)  -> score B-fragments
//   vbuf: [64 hd ][32 s ]  (s contiguous)   -> P@V  B-fragments
__global__ void __launch_bounds__(128)
attn_flash(const _Float16* __restrict__ q, const _Float16* __restrict__ k,
           const _Float16* __restrict__ vt, _Float16* __restrict__ ctx) {
  __shared__ __align__(16) _Float16 kbuf[2][32 * 64];
  __shared__ __align__(16) _Float16 vbuf[2][64 * 32];
  __shared__ __align__(16) _Float16 plds[4][16 * 32];
  const int wave = threadIdx.x >> 5, lane = threadIdx.x & 31;
  const int bh = blockIdx.x >> 5;                     // 64 (b,h) pairs
  const int qt = (blockIdx.x & 31) * 4 + wave;        // 128 q-tiles per bh
  const int b = bh >> 4, h = bh & 15;

  const _Float16* qp = q + (size_t)bh * SS * HD;      // [S,HD]
  const _Float16* kp = k + (size_t)bh * SS * HD;      // [S,HD]
  const _Float16* vp = vt + (size_t)bh * HD * SS;     // [HD,S]

  v8f o[4] = {};
  float mrow[8], lrow[8];
#pragma unroll
  for (int j = 0; j < 8; ++j) { mrow[j] = -3.402823466e38f; lrow[j] = 0.f; }

  const int qrow = qt * 16 + (lane & 15);
  const int hi = (lane & 16) ? 8 : 0;

  // Q fragments are loop-invariant: load once.
  const v16h aq0 = load_fragA(qp, HD, qrow, 0, lane);
  const v16h aq1 = load_fragA(qp, HD, qrow, 32, lane);

  if (wave == 0) {
    tdm_load_2d((unsigned)(uintptr_t)&kbuf[0][0], kp, 64, 32, HD, SS, HD);
    tdm_load_2d((unsigned)(uintptr_t)&vbuf[0][0], vp, 32, 64, SS, HD, SS);
  }

  for (int kv = 0; kv < SS; kv += 32) {
    const int cur = (kv >> 5) & 1;
    if (wave == 0) __builtin_amdgcn_s_wait_tensorcnt(0);   // cur tiles landed
    __syncthreads();
    if (wave == 0 && (kv + 32) < SS) {
      tdm_load_2d((unsigned)(uintptr_t)&kbuf[cur ^ 1][0],
                  kp + (size_t)(kv + 32) * HD, 64, 32, HD, SS, HD);
      tdm_load_2d((unsigned)(uintptr_t)&vbuf[cur ^ 1][0],
                  vp + (kv + 32), 32, 64, SS, HD, SS);
    }

    // scores 16x32 (two 16x16 C tiles), K-dim = HD = 64, from shared kbuf
    v8f s0 = {}, s1 = {};
    s0 = wmma16(aq0, load_fragB(&kbuf[cur][0], 64, (lane & 15), 0, lane), s0);
    s1 = wmma16(aq0, load_fragB(&kbuf[cur][0], 64, 16 + (lane & 15), 0, lane), s1);
    s0 = wmma16(aq1, load_fragB(&kbuf[cur][0], 64, (lane & 15), 32, lane), s0);
    s1 = wmma16(aq1, load_fragB(&kbuf[cur][0], 64, 16 + (lane & 15), 32, lane), s1);

    // online softmax per row (8 rows per lane; 16-lane row groups)
#pragma unroll
    for (int j = 0; j < 8; ++j) {
      float v = fmaxf(s0[j], s1[j]);
#pragma unroll
      for (int off = 8; off; off >>= 1) v = fmaxf(v, __shfl_xor(v, off, 32));
      const float mn = fmaxf(mrow[j], v);
      const float corr = __expf(mrow[j] - mn);
      const float p0 = __expf(s0[j] - mn);
      const float p1 = __expf(s1[j] - mn);
      float ps = p0 + p1;
#pragma unroll
      for (int off = 8; off; off >>= 1) ps += __shfl_xor(ps, off, 32);
      lrow[j] = lrow[j] * corr + ps;
      mrow[j] = mn;
      o[0][j] *= corr; o[1][j] *= corr; o[2][j] *= corr; o[3][j] *= corr;
      const int r = j + hi;
      plds[wave][r * 32 + (lane & 15)]      = (_Float16)p0;
      plds[wave][r * 32 + 16 + (lane & 15)] = (_Float16)p1;
    }
    // re-read P as an A fragment (16x32), accumulate P @ V(32x64)
    v16h ap;
    {
      const _Float16* p = &plds[wave][(lane & 15) * 32 + ((lane & 16) ? 8 : 0)];
      FragU f;
      f.u[0] = *reinterpret_cast<const uint4*>(p);
      f.u[1] = *reinterpret_cast<const uint4*>(p + 16);
      ap = f.h;
    }
#pragma unroll
    for (int t = 0; t < 4; ++t) {
      v16h bvf = load_fragB(&vbuf[cur][0], 32, t * 16 + (lane & 15), 0, lane);
      o[t] = wmma16(ap, bvf, o[t]);
    }
  }

  // normalize and store context as [B,S,D] f16
#pragma unroll
  for (int t = 0; t < 4; ++t) {
    const int col = h * 64 + t * 16 + (lane & 15);
#pragma unroll
    for (int j = 0; j < 8; ++j) {
      const int r = qt * 16 + j + hi;
      ctx[((size_t)(b * SS + r)) * DD + col] = (_Float16)(o[t][j] / lrow[j]);
    }
  }
}

// ---- output projection: out f32 = ctx[B*S,D] @ w_out + b_out ---------------
__global__ void __launch_bounds__(256)
gemm_out(const _Float16* __restrict__ ctx, const _Float16* __restrict__ wt,
         const float* __restrict__ bias, float* __restrict__ out) {
  __shared__ __align__(16) _Float16 bsh[2][64 * 32];
  const int wave = threadIdx.x >> 5, lane = threadIdx.x & 31;
  const int nt = blockIdx.x % 16, mb = blockIdx.x / 16;   // 32 x 16 blocks
  const int rowbase = mb * 256 + wave * 32;
  const int n0 = nt * 64;

  v8f c0[4] = {}, c1[4] = {};
  gemm_mainloop(ctx, wt, bsh, rowbase, n0, DD, wave, lane, c0, c1);

  const int hi = (lane & 16) ? 8 : 0;
#pragma unroll
  for (int t = 0; t < 4; ++t) {
    const int col = n0 + t * 16 + (lane & 15);
    const float bv = bias[col];
#pragma unroll
    for (int g = 0; g < 2; ++g) {
      const v8f* cc = g ? c1 : c0;
#pragma unroll
      for (int j = 0; j < 8; ++j) {
        const int m = rowbase + g * 16 + j + hi;
        out[(size_t)m * DD + col] = cc[t][j] + bv;
      }
    }
  }
}

// ---- host launch -----------------------------------------------------------
extern "C" void kernel_launch(void* const* d_in, const int* in_sizes, int n_in,
                              void* d_out, int out_size, void* d_ws, size_t ws_size,
                              hipStream_t stream) {
  const float* x     = (const float*)d_in[0];   // [B,S,D]
  const float* w_qkv = (const float*)d_in[1];   // [D,3D]
  const float* b_qkv = (const float*)d_in[2];   // [3D]
  const float* w_out = (const float*)d_in[3];   // [D,D]
  const float* b_out = (const float*)d_in[4];   // [D]
  float* out = (float*)d_out;

  char* ws = (char*)d_ws;
  _Float16* x_h    = (_Float16*)(ws);                    // 16 MiB, reused as ctx
  _Float16* wt_qkv = (_Float16*)(ws + 16777216);         // 6 MiB [3D,D]
  _Float16* wt_out = (_Float16*)(ws + 23068672);         // 2 MiB [D,D]
  _Float16* q_h    = (_Float16*)(ws + 25165824);         // 16 MiB [B,H,S,HD]
  _Float16* k_h    = (_Float16*)(ws + 41943040);         // 16 MiB [B,H,S,HD]
  _Float16* v_t    = (_Float16*)(ws + 58720256);         // 16 MiB [B,H,HD,S]
  _Float16* ctx    = x_h;                                // alias (x_h dead after QKV)

  const int nx = BB * SS * DD;                           // 8388608
  cvt_f32_f16<<<nx / 256, 256, 0, stream>>>(x, x_h, nx);
  transpose_cvt<<<(DD * 3 * DD) / 256, 256, 0, stream>>>(w_qkv, wt_qkv, DD, 3 * DD);
  transpose_cvt<<<(DD * DD) / 256, 256, 0, stream>>>(w_out, wt_out, DD, DD);

  // 32 row-blocks x 48 col-blocks, 8 waves/block, 256x64 macro-tile
  gemm_qkv<<<32 * 48, 256, 0, stream>>>(x_h, wt_qkv, b_qkv, q_h, k_h, v_t);

  // 64 (b,h) x 32 blocks x 4 waves = 128 q-tiles per (b,h)
  attn_flash<<<64 * 32, 128, 0, stream>>>(q_h, k_h, v_t, ctx);

  // 32 row-blocks x 16 col-blocks, 8 waves/block
  gemm_out<<<32 * 16, 256, 0, stream>>>(ctx, wt_out, b_out, out);
}
